// VanillaRNNTrajectoryPredictor_7842610283133
// MI455X (gfx1250) — matrix operand order repaired and checked by reading, not compiled, not run on hardware
//
#include <hip/hip_runtime.h>
#include <hip/hip_bf16.h>

// ---------------- problem constants (from reference) ----------------
#define BSZ   4096
#define CARN  32
#define EDIM  512
#define HDIM  512
#define LNUM  2
#define TDEC  60
#define ODIM  2
#define XW    (6 + CARN * 6)   // 198

typedef __bf16 bf16;
typedef __attribute__((ext_vector_type(16))) bf16  v16bf;
typedef __attribute__((ext_vector_type(8)))  bf16  v8bf;
typedef __attribute__((ext_vector_type(8)))  float v8f;

// ---------------- WMMA fragment loaders (16x16x32 bf16, contiguous k = 16*half + i) ----
// A (16x32, MxK): lane 0-15 row M=lane,   K = kk+0..15
//                 lane 16-31 row M=lane-16, K = kk+16..31
__device__ __forceinline__ v16bf load_a_frag(const bf16* __restrict__ base,
                                             int lda, int row, int kk, int half) {
  const bf16* p0 = base + (size_t)row * lda + kk + half * 16;
  v8bf lo = *(const v8bf*)p0;          // 16B load
  v8bf hi = *(const v8bf*)(p0 + 8);    // 16B load
  v16bf a;
#pragma unroll
  for (int i = 0; i < 8; ++i) { a[i] = lo[i]; a[i + 8] = hi[i]; }
  return a;
}

// B (32x16, KxN): lane 0-15 col N=lane, K = kk+0..15; lane 16-31 col N=lane-16,
// K = kk+16..31.  Weight W is (N x K) row-major (we compute inp @ W^T), so each
// lane reads one contiguous 32B K-chunk of row N.
__device__ __forceinline__ v16bf load_b_frag(const bf16* __restrict__ base,
                                             int ldb, int col, int kk, int half) {
  return *(const v16bf*)(base + (size_t)col * ldb + kk + half * 16); // 32B load
}

// ---------------- fused dual-GEMM RNN cell / head kernel -----------------
// out[BSZ x HDIM] (bf16) = act( A1 @ W1^T + (A2? A2 @ W2^T : 0) + bias )
// act: 0 = tanh, 1 = relu.  K = 512 per operand pair.
// 128-thread block = 4 waves (2M x 2N); each wave owns a 32x32 output tile
// (2 A-frags x 2 B-frags -> 4 WMMAs per K-step, 1 frag load/WMMA).
// No divergence before the WMMAs => EXEC all-1s as the ISA requires.
__global__ __launch_bounds__(128) void rnn_gemm_kernel(
    const bf16* __restrict__ A1, const bf16* __restrict__ W1,
    const bf16* __restrict__ A2, const bf16* __restrict__ W2,
    const float* __restrict__ bias, bf16* __restrict__ out, int act) {
  const int tid  = threadIdx.x;
  const int lane = tid & 31;
  const int wid  = tid >> 5;       // 0..3
  const int wm   = wid & 1;        // 2 waves in M
  const int wn   = wid >> 1;       // 2 waves in N
  const int half = lane >> 4;
  const int lr   = lane & 15;
  const int m0   = blockIdx.x * 64 + wm * 32;
  const int n0   = blockIdx.y * 64 + wn * 32;

  v8f acc[2][2] = {{{}, {}}, {{}, {}}};

#pragma unroll 4
  for (int kk = 0; kk < HDIM; kk += 32) {
    v16bf a0 = load_a_frag(A1, HDIM, m0 + lr,      kk, half);
    v16bf a1 = load_a_frag(A1, HDIM, m0 + 16 + lr, kk, half);
    v16bf b0 = load_b_frag(W1, HDIM, n0 + lr,      kk, half);
    v16bf b1 = load_b_frag(W1, HDIM, n0 + 16 + lr, kk, half);
    acc[0][0] = __builtin_amdgcn_wmma_f32_16x16x32_bf16(false, a0, false, b0,
                                                        (short)0, acc[0][0], false, false);
    acc[0][1] = __builtin_amdgcn_wmma_f32_16x16x32_bf16(false, a0, false, b1,
                                                        (short)0, acc[0][1], false, false);
    acc[1][0] = __builtin_amdgcn_wmma_f32_16x16x32_bf16(false, a1, false, b0,
                                                        (short)0, acc[1][0], false, false);
    acc[1][1] = __builtin_amdgcn_wmma_f32_16x16x32_bf16(false, a1, false, b1,
                                                        (short)0, acc[1][1], false, false);
  }
  if (A2 != nullptr) {
#pragma unroll 4
    for (int kk = 0; kk < HDIM; kk += 32) {
      v16bf a0 = load_a_frag(A2, HDIM, m0 + lr,      kk, half);
      v16bf a1 = load_a_frag(A2, HDIM, m0 + 16 + lr, kk, half);
      v16bf b0 = load_b_frag(W2, HDIM, n0 + lr,      kk, half);
      v16bf b1 = load_b_frag(W2, HDIM, n0 + 16 + lr, kk, half);
      acc[0][0] = __builtin_amdgcn_wmma_f32_16x16x32_bf16(false, a0, false, b0,
                                                          (short)0, acc[0][0], false, false);
      acc[0][1] = __builtin_amdgcn_wmma_f32_16x16x32_bf16(false, a0, false, b1,
                                                          (short)0, acc[0][1], false, false);
      acc[1][0] = __builtin_amdgcn_wmma_f32_16x16x32_bf16(false, a1, false, b0,
                                                          (short)0, acc[1][0], false, false);
      acc[1][1] = __builtin_amdgcn_wmma_f32_16x16x32_bf16(false, a1, false, b1,
                                                          (short)0, acc[1][1], false, false);
    }
  }

  // C/D layout: VGPR v, lane<16 -> (M = base+v,   N = nbase+lane)
  //                      lane>=16 -> (M = base+8+v, N = nbase+lane-16)
#pragma unroll
  for (int j = 0; j < 2; ++j) {
    const int col = n0 + 16 * j + lr;
    const float bs = bias ? bias[col] : 0.0f;
#pragma unroll
    for (int i = 0; i < 2; ++i) {
#pragma unroll
      for (int v = 0; v < 8; ++v) {
        const int r = m0 + 16 * i + half * 8 + v;
        float val = acc[i][j][v] + bs;
        val = (act == 0) ? tanhf(val) : fmaxf(val, 0.0f);
        out[(size_t)r * HDIM + col] = (bf16)val;
      }
    }
  }
}

// ---------------- small prep / glue kernels ----------------
__global__ void cvt_f32_bf16_kernel(const float* __restrict__ s,
                                    bf16* __restrict__ d, int n) {
  int i = blockIdx.x * 256 + threadIdx.x;
  if (i < n) d[i] = (bf16)s[i];
}

__global__ void bias_sum_kernel(const float* __restrict__ a,
                                const float* __restrict__ b,
                                float* __restrict__ o, int n) {
  int i = blockIdx.x * 256 + threadIdx.x;
  if (i < n) o[i] = a[i] + b[i];
}

__global__ void zero_bf16_kernel(bf16* __restrict__ p, int n) {
  int i = blockIdx.x * 256 + threadIdx.x;
  if (i < n) p[i] = (bf16)0.0f;
}

// x_t = tanh(x[:, off:off+6] @ W^T + b)   (K = 6, pure VALU)
__global__ void embed_kernel(const float* __restrict__ x,
                             const float* __restrict__ W,
                             const float* __restrict__ bvec,
                             bf16* __restrict__ xt, int off) {
  int idx = blockIdx.x * 256 + threadIdx.x;        // BSZ * EDIM threads
  int b = idx >> 9;
  int e = idx & (EDIM - 1);
  float s = bvec[e];
#pragma unroll
  for (int j = 0; j < 6; ++j)
    s += x[(size_t)b * XW + off + j] * W[e * 6 + j];
  xt[(size_t)b * EDIM + e] = (bf16)tanhf(s);
}

// preds[b, t, :] = tanh(hrelu[b] @ W2^T + b2)   (O = 2, tiny)
__global__ void head2_kernel(const bf16* __restrict__ hrelu,
                             const float* __restrict__ W2,
                             const float* __restrict__ b2,
                             float* __restrict__ out, int t) {
  int idx = blockIdx.x * 256 + threadIdx.x;        // BSZ * ODIM threads
  if (idx >= BSZ * ODIM) return;
  int b = idx >> 1, o = idx & 1;
  float s = b2[o];
  const bf16* hp = hrelu + (size_t)b * HDIM;
  const float* wp = W2 + (size_t)o * HDIM;
#pragma unroll 8
  for (int k = 0; k < HDIM; ++k) s += (float)hp[k] * wp[k];
  out[(size_t)b * (TDEC * ODIM) + t * ODIM + o] = tanhf(s);
}

// ---------------- host orchestration ----------------
extern "C" void kernel_launch(void* const* d_in, const int* in_sizes, int n_in,
                              void* d_out, int out_size, void* d_ws, size_t ws_size,
                              hipStream_t stream) {
  const float* x          = (const float*)d_in[0];
  const float* emb_main_W = (const float*)d_in[1];
  const float* emb_main_b = (const float*)d_in[2];
  const float* emb_veh_W  = (const float*)d_in[3];
  const float* emb_veh_b  = (const float*)d_in[4];
  const float* enc_Wih    = (const float*)d_in[5];
  const float* enc_Whh    = (const float*)d_in[6];
  const float* enc_bih    = (const float*)d_in[7];
  const float* enc_bhh    = (const float*)d_in[8];
  const float* dec_Wih    = (const float*)d_in[9];
  const float* dec_Whh    = (const float*)d_in[10];
  const float* dec_bih    = (const float*)d_in[11];
  const float* dec_bhh    = (const float*)d_in[12];
  const float* head_W1    = (const float*)d_in[13];
  const float* head_b1    = (const float*)d_in[14];
  const float* head_W2    = (const float*)d_in[15];
  const float* head_b2    = (const float*)d_in[16];
  float* out = (float*)d_out;

  // ---- workspace carve-up (all offsets 256B aligned) ----
  size_t off = 0;
  auto carve = [&](size_t bytes) -> char* {
    char* p = (char*)d_ws + off;
    off += (bytes + 255) & ~(size_t)255;
    return p;
  };
  const size_t WSZ = (size_t)HDIM * HDIM;           // 262144 elems per matrix
  const size_t HS  = (size_t)BSZ * HDIM;            // hidden-state elems
  bf16*  wEncWih = (bf16*)carve(LNUM * WSZ * sizeof(bf16));
  bf16*  wEncWhh = (bf16*)carve(LNUM * WSZ * sizeof(bf16));
  bf16*  wDecWih = (bf16*)carve(LNUM * WSZ * sizeof(bf16));
  bf16*  wDecWhh = (bf16*)carve(LNUM * WSZ * sizeof(bf16));
  bf16*  wHead1  = (bf16*)carve(WSZ * sizeof(bf16));
  float* bsumEnc = (float*)carve(LNUM * HDIM * sizeof(float));
  float* bsumDec = (float*)carve(LNUM * HDIM * sizeof(float));
  bf16*  xt      = (bf16*)carve(HS * sizeof(bf16));
  bf16*  hA      = (bf16*)carve(LNUM * HS * sizeof(bf16));
  bf16*  hB      = (bf16*)carve(LNUM * HS * sizeof(bf16));
  bf16*  hrelu   = (bf16*)carve(HS * sizeof(bf16));
  (void)ws_size; (void)in_sizes; (void)n_in; (void)out_size;

  // ---- one-time prep ----
  {
    int n = (int)(LNUM * WSZ);
    int g = (n + 255) / 256;
    cvt_f32_bf16_kernel<<<g, 256, 0, stream>>>(enc_Wih, wEncWih, n);
    cvt_f32_bf16_kernel<<<g, 256, 0, stream>>>(enc_Whh, wEncWhh, n);
    cvt_f32_bf16_kernel<<<g, 256, 0, stream>>>(dec_Wih, wDecWih, n);
    cvt_f32_bf16_kernel<<<g, 256, 0, stream>>>(dec_Whh, wDecWhh, n);
    int n1 = (int)WSZ;
    cvt_f32_bf16_kernel<<<(n1 + 255) / 256, 256, 0, stream>>>(head_W1, wHead1, n1);
    int nb = LNUM * HDIM;
    bias_sum_kernel<<<(nb + 255) / 256, 256, 0, stream>>>(enc_bih, enc_bhh, bsumEnc, nb);
    bias_sum_kernel<<<(nb + 255) / 256, 256, 0, stream>>>(dec_bih, dec_bhh, bsumDec, nb);
    int nz = (int)(LNUM * HS);
    zero_bf16_kernel<<<(nz + 255) / 256, 256, 0, stream>>>(hA, nz);
  }

  const dim3 gemm_grid(BSZ / 64, HDIM / 64);   // 64 x 8 workgroups
  const dim3 gemm_blk(128);                    // 4 waves, 2M x 2N
  bf16* hprev = hA;
  bf16* hnew  = hB;

  // ---- encoder: 33 steps (32 vehicle tokens + main token) ----
  for (int t = 0; t < CARN + 1; ++t) {
    const bool veh = (t < CARN);
    embed_kernel<<<(BSZ * EDIM) / 256, 256, 0, stream>>>(
        x, veh ? emb_veh_W : emb_main_W, veh ? emb_veh_b : emb_main_b,
        xt, veh ? (6 + 6 * t) : 0);
    // layer 0
    rnn_gemm_kernel<<<gemm_grid, gemm_blk, 0, stream>>>(
        xt, wEncWih, hprev, wEncWhh, bsumEnc, hnew, /*tanh*/0);
    // layer 1
    rnn_gemm_kernel<<<gemm_grid, gemm_blk, 0, stream>>>(
        hnew, wEncWih + WSZ, hprev + HS, wEncWhh + WSZ, bsumEnc + HDIM,
        hnew + HS, /*tanh*/0);
    bf16* tmp = hprev; hprev = hnew; hnew = tmp;
  }

  // ---- decoder: 60 steps; input at step t is h_prev[1] (== enc_tops[-1] at t=0) ----
  for (int t = 0; t < TDEC; ++t) {
    // layer 0: inp = hprev[1], recurrent = hprev[0]
    rnn_gemm_kernel<<<gemm_grid, gemm_blk, 0, stream>>>(
        hprev + HS, wDecWih, hprev, wDecWhh, bsumDec, hnew, /*tanh*/0);
    // layer 1: inp = hnew[0], recurrent = hprev[1]
    rnn_gemm_kernel<<<gemm_grid, gemm_blk, 0, stream>>>(
        hnew, wDecWih + WSZ, hprev + HS, wDecWhh + WSZ, bsumDec + HDIM,
        hnew + HS, /*tanh*/0);
    // head part 1: hrelu = relu(top @ W1^T + b1)
    rnn_gemm_kernel<<<gemm_grid, gemm_blk, 0, stream>>>(
        hnew + HS, wHead1, nullptr, nullptr, head_b1, hrelu, /*relu*/1);
    // head part 2: preds[:, t, :] = tanh(hrelu @ W2^T + b2)
    head2_kernel<<<(BSZ * ODIM + 255) / 256, 256, 0, stream>>>(
        hrelu, head_W2, head_b2, out, t);
    bf16* tmp = hprev; hprev = hnew; hnew = tmp;
  }
}
